// EGNNLayer_420906795769
// MI455X (gfx1250) — matrix operand order repaired and checked by reading, not compiled, not run on hardware
//
#include <hip/hip_runtime.h>

#define N_NODES 50000
#define N_EDGES 800000
#define ND 128
#define ED 16
#define HID 256
#define K1 288            // 273 padded to 288 (9 x 32)
#define KN 384            // 128 + 256
#define EPS 1e-8f

typedef __attribute__((ext_vector_type(16))) __bf16 v16bf;
typedef __attribute__((ext_vector_type(8)))  float  v8f;

struct U32x8 { uint4 a, b; };

__device__ __forceinline__ unsigned short f2u16(float f) {
    __bf16 b = (__bf16)f;                     // native v_cvt bf16 on gfx1250
    return __builtin_bit_cast(unsigned short, b);
}

__device__ __forceinline__ unsigned int pack2(float lo, float hi) {
    return (unsigned int)f2u16(lo) | ((unsigned int)f2u16(hi) << 16);
}

__device__ __forceinline__ float silu_f(float v) {
    return v / (1.0f + expf(-v));
}

// A fragment (16x32 bf16, M x K) from row-major LDS tile, stride in elems.
// Lane l (0..15): M=l, K = {k0..k0+7, k0+16..k0+23}; lane l+16: K = {k0+8.., k0+24..}
__device__ __forceinline__ v16bf ld_a(const unsigned short* lds, int stride, int k0, int lane) {
    const int m  = lane & 15;
    const int kb = (lane >> 4) << 3;
    const unsigned short* p = lds + m * stride + k0 + kb;
    U32x8 r;
    r.a = *reinterpret_cast<const uint4*>(p);
    r.b = *reinterpret_cast<const uint4*>(p + 16);
    return __builtin_bit_cast(v16bf, r);
}

// B fragment (32x16 bf16, K x N) from W^T stored [Nout][Kpad] bf16 (row-major).
// Lane l (0..15): N=n0+l, K = {k0..k0+7, k0+16..k0+23}; lane l+16: K = {k0+8.., k0+24..}
__device__ __forceinline__ v16bf ld_b(const unsigned short* wt, int Kpad, int n0, int k0, int lane) {
    const int n  = n0 + (lane & 15);
    const int kb = (lane >> 4) << 3;
    const unsigned short* p = wt + (size_t)n * Kpad + k0 + kb;
    U32x8 r;
    r.a = *reinterpret_cast<const uint4*>(p);
    r.b = *reinterpret_cast<const uint4*>(p + 16);
    return __builtin_bit_cast(v16bf, r);
}

__device__ __forceinline__ v8f wmma_bf16(v16bf a, v16bf b, v8f c) {
    return __builtin_amdgcn_wmma_f32_16x16x32_bf16(
        /*neg_a=*/false, a, /*neg_b=*/false, b,
        /*c_mod=*/(short)0, c, /*reuse_a=*/false, /*reuse_b=*/false);
}

// ---------------------------------------------------------------------------
__global__ __launch_bounds__(256) void egnn_zero(float* p, long n) {
    long i = (long)blockIdx.x * blockDim.x + threadIdx.x;
    if (i < n) p[i] = 0.0f;
}

// W [K, Nc] f32 (row-major)  ->  W^T [Nc, Kpad] bf16 (zero pad K..Kpad)
__global__ __launch_bounds__(256) void egnn_cvt(const float* __restrict__ src,
                                                unsigned short* __restrict__ dst,
                                                int K, int Nc, int Kpad) {
    int idx = blockIdx.x * blockDim.x + threadIdx.x;
    if (idx >= Nc * Kpad) return;
    int n = idx / Kpad, k = idx - n * Kpad;
    dst[idx] = (k < K) ? f2u16(src[(size_t)k * Nc + n]) : (unsigned short)0;
}

// ---------------------------------------------------------------------------
// 32 edges per block; 8 waves; each wave: 2 M-tiles x 2 N-tiles of 16.
__global__ __launch_bounds__(256) void egnn_edge(
    const float* __restrict__ h, const float* __restrict__ x,
    const float* __restrict__ ea, const long long* __restrict__ eidx,
    const unsigned short* __restrict__ We1T, const float* __restrict__ be1,
    const unsigned short* __restrict__ We2T, const float* __restrict__ be2,
    const unsigned short* __restrict__ Wc1T, const float* __restrict__ bc1,
    const float* __restrict__ Wc2,
    float* __restrict__ agg_msg, float* __restrict__ agg_trans) {

    __shared__ unsigned short ein[32 * K1];   // edge_input tile (bf16); later aliased as m_ij
    __shared__ unsigned short mA[32 * HID];   // m tile, bf16
    __shared__ float          wacc[32];       // coord-head dot accumulators
    __shared__ float          sdiff[96];
    __shared__ float          sinvd[32];
    __shared__ int            srow[32];

    unsigned short* mB = ein;                 // alias: ein dead after stage-1 k-loop + barrier

    const int tid  = threadIdx.x;
    const int lane = tid & 31;
    const int wave = tid >> 5;
    const int e    = tid >> 3;     // edge within tile (0..31)
    const int j    = tid & 7;      // helper index (0..7)
    const long long ge = (long long)blockIdx.x * 32 + e;

    const int row = (int)eidx[ge];
    const int col = (int)eidx[(long long)N_EDGES + ge];
    if (j == 0) srow[e] = row;
    if (tid < 32) wacc[tid] = 0.0f;

    // ---- gather edge_input = [h[row](128) | h[col](128) | dist2 | ea(16) | pad(15)]
    {
        const float4* hr = reinterpret_cast<const float4*>(h + (size_t)row * ND + j * 16);
        float4 a = hr[0], b = hr[1], c = hr[2], d = hr[3];
        *reinterpret_cast<uint4*>(&ein[e * K1 + j * 16]) =
            make_uint4(pack2(a.x, a.y), pack2(a.z, a.w), pack2(b.x, b.y), pack2(b.z, b.w));
        *reinterpret_cast<uint4*>(&ein[e * K1 + j * 16 + 8]) =
            make_uint4(pack2(c.x, c.y), pack2(c.z, c.w), pack2(d.x, d.y), pack2(d.z, d.w));
        const float4* hc = reinterpret_cast<const float4*>(h + (size_t)col * ND + j * 16);
        a = hc[0]; b = hc[1]; c = hc[2]; d = hc[3];
        *reinterpret_cast<uint4*>(&ein[e * K1 + ND + j * 16]) =
            make_uint4(pack2(a.x, a.y), pack2(a.z, a.w), pack2(b.x, b.y), pack2(b.z, b.w));
        *reinterpret_cast<uint4*>(&ein[e * K1 + ND + j * 16 + 8]) =
            make_uint4(pack2(c.x, c.y), pack2(c.z, c.w), pack2(d.x, d.y), pack2(d.z, d.w));
    }
    ein[e * K1 + 257 + 2 * j]     = f2u16(ea[(size_t)ge * ED + 2 * j]);
    ein[e * K1 + 257 + 2 * j + 1] = f2u16(ea[(size_t)ge * ED + 2 * j + 1]);
    ein[e * K1 + 273 + j] = 0;
    if (j < 7) ein[e * K1 + 281 + j] = 0;
    if (j == 0) {
        float dx = x[(size_t)row * 3 + 0] - x[(size_t)col * 3 + 0];
        float dy = x[(size_t)row * 3 + 1] - x[(size_t)col * 3 + 1];
        float dz = x[(size_t)row * 3 + 2] - x[(size_t)col * 3 + 2];
        float d2 = dx * dx + dy * dy + dz * dz;
        ein[e * K1 + 256] = f2u16(d2);
        sdiff[e * 3 + 0] = dx; sdiff[e * 3 + 1] = dy; sdiff[e * 3 + 2] = dz;
        sinvd[e] = 1.0f / (sqrtf(d2 + EPS) + EPS);
    }
    __syncthreads();

    const int n0 = wave * 32;           // each wave: N-tiles n0, n0+16
    const int nn = lane & 15;
    const int mbase = (lane >> 4) * 8;

    // ---- stage 1: m = silu(edge_input @ W_e1 + b_e1)   K = 288
    {
        v8f a00 = {0,0,0,0,0,0,0,0}, a01 = {0,0,0,0,0,0,0,0};
        v8f a10 = {0,0,0,0,0,0,0,0}, a11 = {0,0,0,0,0,0,0,0};
        for (int k0 = 0; k0 < K1; k0 += 32) {
            v16bf fa0 = ld_a(ein,            K1, k0, lane);
            v16bf fa1 = ld_a(ein + 16 * K1,  K1, k0, lane);
            v16bf fb0 = ld_b(We1T, K1, n0,      k0, lane);
            v16bf fb1 = ld_b(We1T, K1, n0 + 16, k0, lane);
            a00 = wmma_bf16(fa0, fb0, a00);
            a01 = wmma_bf16(fa0, fb1, a01);
            a10 = wmma_bf16(fa1, fb0, a10);
            a11 = wmma_bf16(fa1, fb1, a11);
        }
        float bv0 = be1[n0 + nn], bv1 = be1[n0 + 16 + nn];
        #pragma unroll
        for (int r = 0; r < 8; r++) {
            int m0 = mbase + r, m1 = 16 + mbase + r;
            mA[m0 * HID + n0 + nn]      = f2u16(silu_f(a00[r] + bv0));
            mA[m0 * HID + n0 + 16 + nn] = f2u16(silu_f(a01[r] + bv1));
            mA[m1 * HID + n0 + nn]      = f2u16(silu_f(a10[r] + bv0));
            mA[m1 * HID + n0 + 16 + nn] = f2u16(silu_f(a11[r] + bv1));
        }
    }
    __syncthreads();

    // ---- stage 2: m_ij = silu(m @ W_e2 + b_e2); scatter-add into agg_msg
    {
        v8f a00 = {0,0,0,0,0,0,0,0}, a01 = {0,0,0,0,0,0,0,0};
        v8f a10 = {0,0,0,0,0,0,0,0}, a11 = {0,0,0,0,0,0,0,0};
        for (int k0 = 0; k0 < HID; k0 += 32) {
            v16bf fa0 = ld_a(mA,             HID, k0, lane);
            v16bf fa1 = ld_a(mA + 16 * HID,  HID, k0, lane);
            v16bf fb0 = ld_b(We2T, HID, n0,      k0, lane);
            v16bf fb1 = ld_b(We2T, HID, n0 + 16, k0, lane);
            a00 = wmma_bf16(fa0, fb0, a00);
            a01 = wmma_bf16(fa0, fb1, a01);
            a10 = wmma_bf16(fa1, fb0, a10);
            a11 = wmma_bf16(fa1, fb1, a11);
        }
        float bv0 = be2[n0 + nn], bv1 = be2[n0 + 16 + nn];
        #pragma unroll
        for (int r = 0; r < 8; r++) {
            int m0 = mbase + r, m1 = 16 + mbase + r;
            float v00 = silu_f(a00[r] + bv0);
            float v01 = silu_f(a01[r] + bv1);
            float v10 = silu_f(a10[r] + bv0);
            float v11 = silu_f(a11[r] + bv1);
            mB[m0 * HID + n0 + nn]      = f2u16(v00);
            mB[m0 * HID + n0 + 16 + nn] = f2u16(v01);
            mB[m1 * HID + n0 + nn]      = f2u16(v10);
            mB[m1 * HID + n0 + 16 + nn] = f2u16(v11);
            size_t b0a = (size_t)srow[m0] * HID;
            size_t b1a = (size_t)srow[m1] * HID;
            atomicAdd(&agg_msg[b0a + n0 + nn],      v00);
            atomicAdd(&agg_msg[b0a + n0 + 16 + nn], v01);
            atomicAdd(&agg_msg[b1a + n0 + nn],      v10);
            atomicAdd(&agg_msg[b1a + n0 + 16 + nn], v11);
        }
    }
    __syncthreads();

    // ---- stage 3: c = silu(m_ij @ W_c1 + b_c1); fold dot with W_c2 into LDS atomics
    {
        v8f a00 = {0,0,0,0,0,0,0,0}, a01 = {0,0,0,0,0,0,0,0};
        v8f a10 = {0,0,0,0,0,0,0,0}, a11 = {0,0,0,0,0,0,0,0};
        for (int k0 = 0; k0 < HID; k0 += 32) {
            v16bf fa0 = ld_a(mB,             HID, k0, lane);
            v16bf fa1 = ld_a(mB + 16 * HID,  HID, k0, lane);
            v16bf fb0 = ld_b(Wc1T, HID, n0,      k0, lane);
            v16bf fb1 = ld_b(Wc1T, HID, n0 + 16, k0, lane);
            a00 = wmma_bf16(fa0, fb0, a00);
            a01 = wmma_bf16(fa0, fb1, a01);
            a10 = wmma_bf16(fa1, fb0, a10);
            a11 = wmma_bf16(fa1, fb1, a11);
        }
        float bv0 = bc1[n0 + nn],  bv1 = bc1[n0 + 16 + nn];
        float w0  = Wc2[n0 + nn],  w1  = Wc2[n0 + 16 + nn];
        #pragma unroll
        for (int r = 0; r < 8; r++) {
            int m0 = mbase + r, m1 = 16 + mbase + r;
            atomicAdd(&wacc[m0], silu_f(a00[r] + bv0) * w0 + silu_f(a01[r] + bv1) * w1);
            atomicAdd(&wacc[m1], silu_f(a10[r] + bv0) * w0 + silu_f(a11[r] + bv1) * w1);
        }
    }
    __syncthreads();

    // ---- w = tanh(wacc); trans scatter-add
    if (tid < 32) {
        float f = tanhf(wacc[tid]) * sinvd[tid];
        size_t base = (size_t)srow[tid] * 3;
        atomicAdd(&agg_trans[base + 0], sdiff[tid * 3 + 0] * f);
        atomicAdd(&agg_trans[base + 1], sdiff[tid * 3 + 1] * f);
        atomicAdd(&agg_trans[base + 2], sdiff[tid * 3 + 2] * f);
    }
}

// ---------------------------------------------------------------------------
__global__ __launch_bounds__(256) void egnn_node(
    const float* __restrict__ h, const float* __restrict__ x,
    const float* __restrict__ nmask,
    const unsigned short* __restrict__ Wn1T, const float* __restrict__ bn1,
    const unsigned short* __restrict__ Wn2T, const float* __restrict__ bn2,
    const float* __restrict__ agg_msg, const float* __restrict__ agg_trans,
    float* __restrict__ out_h, float* __restrict__ out_x) {

    __shared__ unsigned short nin[16 * KN];   // [h | agg_msg] tile, bf16
    __shared__ unsigned short tt[16 * HID];   // hidden tile, bf16

    const int tid  = threadIdx.x;
    const int lane = tid & 31;
    const int wave = tid >> 5;
    const int e    = tid >> 4;
    const int j    = tid & 15;
    const int gmb  = blockIdx.x * 16;

    // gather node_input = [h(128) | agg_msg(256)]
    {
        const float4* hp = reinterpret_cast<const float4*>(h + (size_t)(gmb + e) * ND + j * 8);
        float4 a = hp[0], b = hp[1];
        *reinterpret_cast<uint4*>(&nin[e * KN + j * 8]) =
            make_uint4(pack2(a.x, a.y), pack2(a.z, a.w), pack2(b.x, b.y), pack2(b.z, b.w));
        const float4* ap = reinterpret_cast<const float4*>(agg_msg + (size_t)(gmb + e) * HID + j * 16);
        float4 c0 = ap[0], c1 = ap[1], c2 = ap[2], c3 = ap[3];
        *reinterpret_cast<uint4*>(&nin[e * KN + ND + j * 16]) =
            make_uint4(pack2(c0.x, c0.y), pack2(c0.z, c0.w), pack2(c1.x, c1.y), pack2(c1.z, c1.w));
        *reinterpret_cast<uint4*>(&nin[e * KN + ND + j * 16 + 8]) =
            make_uint4(pack2(c2.x, c2.y), pack2(c2.z, c2.w), pack2(c3.x, c3.y), pack2(c3.z, c3.w));
    }
    __syncthreads();

    const int nn = lane & 15;
    const int mbase = (lane >> 4) * 8;

    // ---- t = silu(node_input @ W_n1 + b_n1)   K = 384
    {
        const int n0 = wave * 32;
        v8f acc0 = {0,0,0,0,0,0,0,0}, acc1 = {0,0,0,0,0,0,0,0};
        for (int k0 = 0; k0 < KN; k0 += 32) {
            v16bf a  = ld_a(nin, KN, k0, lane);
            v16bf b0 = ld_b(Wn1T, KN, n0,      k0, lane);
            v16bf b1 = ld_b(Wn1T, KN, n0 + 16, k0, lane);
            acc0 = wmma_bf16(a, b0, acc0);
            acc1 = wmma_bf16(a, b1, acc1);
        }
        float bv0 = bn1[n0 + nn], bv1 = bn1[n0 + 16 + nn];
        #pragma unroll
        for (int r = 0; r < 8; r++) {
            int m = mbase + r;
            tt[m * HID + n0 + nn]      = f2u16(silu_f(acc0[r] + bv0));
            tt[m * HID + n0 + 16 + nn] = f2u16(silu_f(acc1[r] + bv1));
        }
    }
    __syncthreads();

    // ---- h_new = h + t @ W_n2 + b_n2   (128 out cols: 1 N-tile per wave)
    {
        const int n0 = wave * 16;
        v8f acc = {0,0,0,0,0,0,0,0};
        for (int k0 = 0; k0 < HID; k0 += 32) {
            v16bf a = ld_a(tt, HID, k0, lane);
            v16bf b = ld_b(Wn2T, HID, n0, k0, lane);
            acc = wmma_bf16(a, b, acc);
        }
        float bv = bn2[n0 + nn];
        #pragma unroll
        for (int r = 0; r < 8; r++) {
            int m = mbase + r;
            size_t idx = (size_t)(gmb + m) * ND + n0 + nn;
            out_h[idx] = h[idx] + acc[r] + bv;
        }
    }

    // ---- x_new = x + agg_trans * node_mask
    if (tid < 16) {
        int gm = gmb + tid;
        float msk = nmask[gm];
        #pragma unroll
        for (int c = 0; c < 3; c++) {
            size_t idx = (size_t)gm * 3 + c;
            out_x[idx] = x[idx] + agg_trans[idx] * msk;
        }
    }
}

// ---------------------------------------------------------------------------
extern "C" void kernel_launch(void* const* d_in, const int* in_sizes, int n_in,
                              void* d_out, int out_size, void* d_ws, size_t ws_size,
                              hipStream_t stream) {
    (void)in_sizes; (void)n_in; (void)out_size; (void)ws_size;
    const float*     h     = (const float*)d_in[0];
    const float*     x     = (const float*)d_in[1];
    const float*     ea    = (const float*)d_in[2];
    const float*     nmask = (const float*)d_in[3];
    const float*     We1   = (const float*)d_in[4];
    const float*     be1   = (const float*)d_in[5];
    const float*     We2   = (const float*)d_in[6];
    const float*     be2   = (const float*)d_in[7];
    const float*     Wc1   = (const float*)d_in[8];
    const float*     bc1   = (const float*)d_in[9];
    const float*     Wc2   = (const float*)d_in[10];
    const float*     Wn1   = (const float*)d_in[11];
    const float*     bn1   = (const float*)d_in[12];
    const float*     Wn2   = (const float*)d_in[13];
    const float*     bn2   = (const float*)d_in[14];
    const long long* eidx  = (const long long*)d_in[15];

    char* ws = (char*)d_ws;
    float* agg_msg   = (float*)ws;                                   // N*256 f32
    float* agg_trans = (float*)(ws + (size_t)N_NODES * HID * 4);     // N*3   f32 (contiguous)
    size_t off = (size_t)N_NODES * (HID + 3) * 4;
    unsigned short* We1T = (unsigned short*)(ws + off); off += (size_t)HID * K1  * 2;
    unsigned short* We2T = (unsigned short*)(ws + off); off += (size_t)HID * HID * 2;
    unsigned short* Wc1T = (unsigned short*)(ws + off); off += (size_t)HID * HID * 2;
    unsigned short* Wn1T = (unsigned short*)(ws + off); off += (size_t)HID * KN  * 2;
    unsigned short* Wn2T = (unsigned short*)(ws + off); off += (size_t)ND  * HID * 2;

    long zn = (long)N_NODES * (HID + 3);
    egnn_zero<<<(unsigned)((zn + 255) / 256), 256, 0, stream>>>(agg_msg, zn);

    egnn_cvt<<<(HID * K1  + 255) / 256, 256, 0, stream>>>(We1, We1T, 273, HID, K1);
    egnn_cvt<<<(HID * HID + 255) / 256, 256, 0, stream>>>(We2, We2T, HID, HID, HID);
    egnn_cvt<<<(HID * HID + 255) / 256, 256, 0, stream>>>(Wc1, Wc1T, HID, HID, HID);
    egnn_cvt<<<(HID * KN  + 255) / 256, 256, 0, stream>>>(Wn1, Wn1T, KN,  HID, KN);
    egnn_cvt<<<(ND  * HID + 255) / 256, 256, 0, stream>>>(Wn2, Wn2T, HID, ND,  HID);

    egnn_edge<<<N_EDGES / 32, 256, 0, stream>>>(h, x, ea, eidx,
                                                We1T, be1, We2T, be2, Wc1T, bc1, Wc2,
                                                agg_msg, agg_trans);

    egnn_node<<<N_NODES / 16, 256, 0, stream>>>(h, x, nmask,
                                                Wn1T, bn1, Wn2T, bn2,
                                                agg_msg, agg_trans,
                                                (float*)d_out,
                                                (float*)d_out + (size_t)N_NODES * ND);
}
